// AttentionHead_77910706749666
// MI455X (gfx1250) — matrix operand order repaired and checked by reading, not compiled, not run on hardware
//
#include <hip/hip_runtime.h>

#define BB 8
#define SS 2048
#define DD 512
#define NN (BB * SS)   // 16384 rows

typedef __attribute__((ext_vector_type(16))) __bf16 v16bf;
typedef __attribute__((ext_vector_type(8)))  float  v8f;
typedef __attribute__((ext_vector_type(16))) unsigned short v16u;
typedef __attribute__((ext_vector_type(8)))  unsigned short v8u;
typedef __attribute__((ext_vector_type(4)))  unsigned int u32x4;
typedef __attribute__((ext_vector_type(8)))  int i32x8;
typedef __attribute__((ext_vector_type(4)))  int i32x4;

#if __has_builtin(__builtin_amdgcn_tensor_load_to_lds) && \
    __has_builtin(__builtin_amdgcn_s_wait_tensorcnt)
#define HAVE_TDM 1
#else
#define HAVE_TDM 0
#endif

static __device__ __forceinline__ unsigned short f2bf(float f) {
  unsigned int u = __float_as_uint(f);
  u += 0x7FFFu + ((u >> 16) & 1u);   // round-to-nearest-even
  return (unsigned short)(u >> 16);
}

static __device__ __forceinline__ v16bf as_bf(v16u x) {
  return __builtin_bit_cast(v16bf, x);
}

static __device__ __forceinline__ v16u cat8(v8u lo, v8u hi) {
  return __builtin_shufflevector(lo, hi, 0, 1, 2, 3, 4, 5, 6, 7,
                                          8, 9, 10, 11, 12, 13, 14, 15);
}

#if HAVE_TDM
// 2-D TDM load: tile_h lines x tile_w bf16 elements (line stride stride_elems)
// from global into LDS at lds_off.  tensor dims sized to the tile (no OOB).
static __device__ __forceinline__ void tdm_load_2d(unsigned lds_off,
                                                   const void* gaddr,
                                                   unsigned tile_w,
                                                   unsigned tile_h,
                                                   unsigned stride_elems) {
  unsigned long long ga = (unsigned long long)gaddr;
  u32x4 g0;
  g0[0] = 1u;                                   // count=1, user descriptor
  g0[1] = lds_off;                              // lds_addr (bytes)
  g0[2] = (unsigned)(ga & 0xFFFFFFFFull);       // global_addr[31:0]
  g0[3] = (unsigned)((ga >> 32) & 0x01FFFFFFull) | (2u << 30);  // [56:32]+type=2

  i32x8 g1;
  g1[0] = (int)(1u << 16);                      // data_size=1 (2 bytes/elem)
  g1[1] = (int)(tile_w << 16);                  // tensor_dim0[15:0] @ bits 63:48
  g1[2] = (int)((tile_w >> 16) | (tile_h << 16));  // dim0 hi | tensor_dim1 lo
  g1[3] = (int)((tile_h >> 16) | (tile_w << 16));  // dim1 hi | tile_dim0
  g1[4] = (int)(tile_h);                        // tile_dim1 (tile_dim2=0)
  g1[5] = (int)(stride_elems);                  // tensor_dim0_stride[31:0]
  g1[6] = 0;                                    // stride hi | dim1_stride lo
  g1[7] = 0;                                    // dim1_stride hi

  i32x4 gz = {0, 0, 0, 0};                      // 2-D tensor: groups 2/3 unused
#if __clang_major__ >= 23
  i32x8 gz8 = {0, 0, 0, 0, 0, 0, 0, 0};
  __builtin_amdgcn_tensor_load_to_lds(g0, g1, gz, gz, gz8, 0);
#else
  __builtin_amdgcn_tensor_load_to_lds(g0, g1, gz, gz, 0);
#endif
}
#endif

// ---------------------------------------------------------------- convert
__global__ void cvt_f32_bf16(const float* __restrict__ src,
                             unsigned short* __restrict__ dst, int n) {
  int i = blockIdx.x * blockDim.x + threadIdx.x;
  int stride = gridDim.x * blockDim.x;
  for (; i < n; i += stride) dst[i] = f2bf(src[i]);
}

// ---------------------------------------------------------------- QKV proj
// out = X @ W^T.  Q/K stored row-major [b*S+s][e]; V stored transposed
// [b][e][s] so the attention PV B-operand is contiguous.
// 64x512 weight tile staged in LDS (TDM), shared by all 8 waves.
__global__ void qkv_proj(const unsigned short* __restrict__ Xb,
                         const unsigned short* __restrict__ Wqb,
                         const unsigned short* __restrict__ Wkb,
                         const unsigned short* __restrict__ Wvb,
                         unsigned short* __restrict__ Qb,
                         unsigned short* __restrict__ Kb,
                         unsigned short* __restrict__ Vtb) {
  __shared__ unsigned short Wsh[64 * DD];       // 64 KB
  const int tid  = threadIdx.x;
  const int lane = tid & 31;
  const int wave = tid >> 5;
  const int half = lane >> 4;
  const int l16  = lane & 15;
  const int r0 = blockIdx.x * 128 + wave * 16;
  const int e0 = blockIdx.y * 64;
  const unsigned short* W = (blockIdx.z == 0) ? Wqb : (blockIdx.z == 1) ? Wkb : Wvb;

#if HAVE_TDM
  if (wave == 0) {
    tdm_load_2d((unsigned)(size_t)(void*)Wsh, W + (size_t)e0 * DD, DD, 64, DD);
    __builtin_amdgcn_s_wait_tensorcnt(0);
  }
#else
#pragma unroll
  for (int i = 0; i < 16; ++i) {
    int off = (tid + i * 256) * 8;       // 256 thr x 16 x 8 ushort = 64x512
    *(v8u*)(Wsh + off) = *(const v8u*)(W + (size_t)e0 * DD + off);
  }
#endif
  __syncthreads();

  v8f acc[4];
#pragma unroll
  for (int t = 0; t < 4; ++t) acc[t] = (v8f){0.f,0.f,0.f,0.f,0.f,0.f,0.f,0.f};

  for (int d0 = 0; d0 < DD; d0 += 32) {
    // A operand: X rows (lanes 0-15: K {0..15}; lanes 16-31: {16..31})
    const unsigned short* ap = Xb + (size_t)(r0 + l16) * DD + d0;
    v16bf A = as_bf(cat8(*(const v8u*)(ap + 16 * half),
                         *(const v8u*)(ap + 16 * half + 8)));
#pragma unroll
    for (int t = 0; t < 4; ++t) {
      // B operand: B[kk,n] = W[e0+n, d0+kk]; lane holds 16 contiguous d-values
      const unsigned short* bp = Wsh + (size_t)(t * 16 + l16) * DD + d0 + 16 * half;
      v16bf Bm = as_bf(cat8(*(const v8u*)bp, *(const v8u*)(bp + 8)));
      acc[t] = __builtin_amdgcn_wmma_f32_16x16x32_bf16(
          false, A, false, Bm, (short)0, acc[t], false, false);
    }
  }
  // C layout: VGPR j -> M = j + 8*half, col = n = l16
  if (blockIdx.z < 2) {
    unsigned short* Dst = (blockIdx.z == 0) ? Qb : Kb;
#pragma unroll
    for (int t = 0; t < 4; ++t)
#pragma unroll
      for (int j = 0; j < 8; ++j) {
        int row = r0 + j + 8 * half;
        int col = e0 + t * 16 + l16;
        Dst[(size_t)row * DD + col] = f2bf(acc[t][j]);
      }
  } else {
    const int bidx = r0 >> 11;           // 128-row blocks never straddle a batch
    const int srow = r0 & (SS - 1);
#pragma unroll
    for (int t = 0; t < 4; ++t)
#pragma unroll
      for (int j = 0; j < 8; ++j) {
        int s = srow + j + 8 * half;
        int col = e0 + t * 16 + l16;
        Vtb[((size_t)bidx * DD + col) * SS + s] = f2bf(acc[t][j]);
      }
  }
}

// ---------------------------------------------------------------- attention
// Block: 64 q-rows, 8 waves = 4 row-tiles x 2 D-halves. Flash over 32-key
// blocks, double-buffered TDM staging of K (row-major) and Vt tiles.
__global__ void flash_attn(const unsigned short* __restrict__ Qb,
                           const unsigned short* __restrict__ Kb,
                           const unsigned short* __restrict__ Vtb,
                           const int* __restrict__ mask,
                           float* __restrict__ out) {
  __shared__ unsigned short Qs[64 * DD];        // 64 KB
  __shared__ unsigned short Ks[2][32 * DD];     // 2 x 32 KB  [key][d]
  __shared__ unsigned short Vts[2][DD * 32];    // 2 x 32 KB  [e][key]
  __shared__ unsigned short Ps[8][16 * 32];     // 8 KB, per-wave P staging

  const int tid  = threadIdx.x;
  const int lane = tid & 31;
  const int wave = tid >> 5;
  const int half = lane >> 4;
  const int l16  = lane & 15;
  const int b    = blockIdx.y;
  const int qrl  = (wave & 3) * 16;                    // local q-row of this wave
  const int q0   = blockIdx.x * 64 + qrl;              // row within sequence
  const int ebase = (wave >> 2) * 256;                 // this wave's D half
  const size_t qblk = ((size_t)b * SS + blockIdx.x * 64) * DD;

  unsigned short* Pw = &Ps[wave][0];

  // ---- stage the block's 64x512 Q tile into LDS (once) + first K/V block
#if HAVE_TDM
  if (wave == 0) {
    tdm_load_2d((unsigned)(size_t)(void*)Qs, Qb + qblk, DD, 64, DD);
    tdm_load_2d((unsigned)(size_t)(void*)&Ks[0][0],
                Kb + (size_t)b * SS * DD, DD, 32, DD);
    tdm_load_2d((unsigned)(size_t)(void*)&Vts[0][0],
                Vtb + (size_t)b * DD * SS, 32, DD, SS);
  }
#else
#pragma unroll
  for (int i = 0; i < 16; ++i) {
    int off = (tid + i * 256) * 8;       // 256 thr x 16 x 8 ushort = 64x512
    *(v8u*)(Qs + off) = *(const v8u*)(Qb + qblk + off);
  }
#endif

  v8f o[16];
#pragma unroll
  for (int t = 0; t < 16; ++t) o[t] = (v8f){0.f,0.f,0.f,0.f,0.f,0.f,0.f,0.f};
  float mrow[8], lrow[8];
#pragma unroll
  for (int j = 0; j < 8; ++j) { mrow[j] = -1e30f; lrow[j] = 0.f; }

  const float scale = 0.022097086912079608f;  // 1/sqrt(S), S=2048

  for (int k0 = 0; k0 < SS; k0 += 32) {
    const int buf = (k0 >> 5) & 1;
    __syncthreads();  // all waves done reading the buffer we are about to fill
#if HAVE_TDM
    if (wave == 0) {
      if (k0 + 32 < SS) {   // pipeline: issue next block, then wait current
        const size_t kb2 = ((size_t)b * SS + k0 + 32) * DD;
        const size_t vb2 = (size_t)b * DD * SS + k0 + 32;
        tdm_load_2d((unsigned)(size_t)(void*)&Ks[buf ^ 1][0], Kb + kb2, DD, 32, DD);
        tdm_load_2d((unsigned)(size_t)(void*)&Vts[buf ^ 1][0], Vtb + vb2, 32, DD, SS);
        __builtin_amdgcn_s_wait_tensorcnt(2);   // all but the 2 just issued
      } else {
        __builtin_amdgcn_s_wait_tensorcnt(0);
      }
    }
#else
    {
      const size_t kbase  = ((size_t)b * SS + k0) * DD;
      const size_t vtbase = (size_t)b * DD * SS + k0;
#pragma unroll
      for (int i = 0; i < 8; ++i) {
        int off = (tid + i * 256) * 8;   // 32x512 K tile
        *(v8u*)(&Ks[buf][0] + off) = *(const v8u*)(Kb + kbase + off);
      }
#pragma unroll
      for (int i = 0; i < 8; ++i) {      // 512x32 Vt tile (64-byte lines)
        int c = tid + i * 256;           // 2048 chunks of 8 ushorts
        int e = c >> 2, kk = (c & 3) * 8;
        *(v8u*)(&Vts[buf][0] + e * 32 + kk) =
            *(const v8u*)(Vtb + vtbase + (size_t)e * SS + kk);
      }
    }
#endif
    __syncthreads();

    const unsigned short* KsB  = &Ks[buf][0];
    const unsigned short* VtsB = &Vts[buf][0];

    // ---- scores: S[m,n] = sum_d Q[q0+m,d] * K[k0+n,d]   (all LDS operands)
    v8f s0 = (v8f){0.f,0.f,0.f,0.f,0.f,0.f,0.f,0.f};
    v8f s1 = (v8f){0.f,0.f,0.f,0.f,0.f,0.f,0.f,0.f};
    for (int d0 = 0; d0 < DD; d0 += 32) {
      const unsigned short* ap = Qs + (size_t)(qrl + l16) * DD + d0;
      v16bf A = as_bf(cat8(*(const v8u*)(ap + 16 * half),
                           *(const v8u*)(ap + 16 * half + 8)));
      const unsigned short* kp0 = KsB + (size_t)l16 * DD + d0 + 16 * half;
      const unsigned short* kp1 = KsB + (size_t)(16 + l16) * DD + d0 + 16 * half;
      v16bf B0 = as_bf(cat8(*(const v8u*)kp0, *(const v8u*)(kp0 + 8)));
      v16bf B1 = as_bf(cat8(*(const v8u*)kp1, *(const v8u*)(kp1 + 8)));
      s0 = __builtin_amdgcn_wmma_f32_16x16x32_bf16(false, A, false, B0, (short)0, s0, false, false);
      s1 = __builtin_amdgcn_wmma_f32_16x16x32_bf16(false, A, false, B1, (short)0, s1, false, false);
    }

    // ---- mask + scale (each lane's column = one key)
    const float NEG = -1e30f;
    const int ma = mask[b * SS + k0 + l16];
    const int mb = mask[b * SS + k0 + 16 + l16];
#pragma unroll
    for (int j = 0; j < 8; ++j) {
      s0[j] = ma ? s0[j] * scale : NEG;
      s1[j] = mb ? s1[j] * scale : NEG;
    }

    // ---- online softmax (rows split across 16-lane halves; reduce with xor<=8)
    float p0[8], p1[8], corr[8];
#pragma unroll
    for (int j = 0; j < 8; ++j) {
      float t = fmaxf(s0[j], s1[j]);
      t = fmaxf(t, __shfl_xor(t, 1, 32));
      t = fmaxf(t, __shfl_xor(t, 2, 32));
      t = fmaxf(t, __shfl_xor(t, 4, 32));
      t = fmaxf(t, __shfl_xor(t, 8, 32));
      float mn = fmaxf(mrow[j], t);
      corr[j] = __expf(mrow[j] - mn);
      mrow[j] = mn;
      p0[j] = __expf(s0[j] - mn);
      p1[j] = __expf(s1[j] - mn);
      float sum = p0[j] + p1[j];
      sum += __shfl_xor(sum, 1, 32);
      sum += __shfl_xor(sum, 2, 32);
      sum += __shfl_xor(sum, 4, 32);
      sum += __shfl_xor(sum, 8, 32);
      lrow[j] = lrow[j] * corr[j] + sum;
    }
#pragma unroll
    for (int t = 0; t < 16; ++t)
#pragma unroll
      for (int j = 0; j < 8; ++j) o[t][j] *= corr[j];

    // ---- P (C layout) -> LDS -> reload in A layout
#pragma unroll
    for (int j = 0; j < 8; ++j) {
      int row = j + 8 * half;
      Pw[row * 32 + l16]      = f2bf(p0[j]);
      Pw[row * 32 + 16 + l16] = f2bf(p1[j]);
    }
    v16bf P = as_bf(cat8(*(const v8u*)(Pw + l16 * 32 + 16 * half),
                         *(const v8u*)(Pw + l16 * 32 + 16 * half + 8)));

    // ---- O += P @ V: B[kk,n] = Vts[e0+n][kk] -- contiguous 16-half loads
#pragma unroll
    for (int t = 0; t < 16; ++t) {
      int e0 = ebase + t * 16;
      const unsigned short* vp = VtsB + (size_t)(e0 + l16) * 32 + 16 * half;
      v16bf Bv = as_bf(cat8(*(const v8u*)vp, *(const v8u*)(vp + 8)));
      o[t] = __builtin_amdgcn_wmma_f32_16x16x32_bf16(
          false, P, false, Bv, (short)0, o[t], false, false);
    }
  }

  // ---- epilogue: out = O / l  (fp32)
#pragma unroll
  for (int t = 0; t < 16; ++t) {
    int e0 = ebase + t * 16;
#pragma unroll
    for (int j = 0; j < 8; ++j) {
      int row = q0 + j + 8 * half;
      out[((size_t)b * SS + row) * DD + e0 + l16] = o[t][j] / lrow[j];
    }
  }
}

// ---------------------------------------------------------------- launch
extern "C" void kernel_launch(void* const* d_in, const int* in_sizes, int n_in,
                              void* d_out, int out_size, void* d_ws, size_t ws_size,
                              hipStream_t stream) {
  const float* X   = (const float*)d_in[0];
  const int*  mask = (const int*)d_in[1];
  const float* Wq  = (const float*)d_in[2];
  const float* Wk  = (const float*)d_in[3];
  const float* Wv  = (const float*)d_in[4];
  float* out = (float*)d_out;

  const size_t ND = (size_t)NN * DD;          // 8,388,608
  const size_t WD = (size_t)DD * DD;          // 262,144
  const size_t need = (4 * ND + 3 * WD) * sizeof(unsigned short);
  if (ws_size < need) return;

  unsigned short* ws  = (unsigned short*)d_ws;
  unsigned short* Xb  = ws;
  unsigned short* Qb  = ws + ND;
  unsigned short* Kb  = ws + 2 * ND;
  unsigned short* Vtb = ws + 3 * ND;          // transposed V: [b][e][s]
  unsigned short* Wqb = ws + 4 * ND;
  unsigned short* Wkb = Wqb + WD;
  unsigned short* Wvb = Wkb + WD;

  cvt_f32_bf16<<<2048, 256, 0, stream>>>(X, Xb, (int)ND);
  cvt_f32_bf16<<<256, 256, 0, stream>>>(Wq, Wqb, (int)WD);
  cvt_f32_bf16<<<256, 256, 0, stream>>>(Wk, Wkb, (int)WD);
  cvt_f32_bf16<<<256, 256, 0, stream>>>(Wv, Wvb, (int)WD);

  qkv_proj<<<dim3(NN / 128, DD / 64, 3), 256, 0, stream>>>(
      Xb, Wqb, Wkb, Wvb, Qb, Kb, Vtb);

  flash_attn<<<dim3(SS / 64, BB), 256, 0, stream>>>(Qb, Kb, Vtb, mask, out);
}
